// MultiHeadAttention_69303592289013
// MI455X (gfx1250) — matrix operand order repaired and checked by reading, not compiled, not run on hardware
//
#include <hip/hip_runtime.h>
#include <hip/hip_bf16.h>

// MHA forward, CDNA5 wave32 WMMA implementation (register-blocked).
// Pipeline: cvt f32->f16, QKV GEMM 64x64/wave (-> Q*0.125, K, V^T),
// flash attention 2 q-tiles/wave (S^T = K Q^T trick keeps P in A-fragment
// layout in-lane), out-proj GEMM 64x64/wave.

typedef _Float16 half_t;
typedef half_t v16h __attribute__((ext_vector_type(16)));
typedef half_t v8h  __attribute__((ext_vector_type(8)));
typedef half_t v4h  __attribute__((ext_vector_type(4)));
typedef float  v8f  __attribute__((ext_vector_type(8)));

static constexpr int Bq = 4, Sq = 2048, DIN = 1024, EMB = 1024, Hh = 16, HD = 64;

__device__ __forceinline__ v8f wmma_f16(v16h a, v16h b, v8f c) {
  return __builtin_amdgcn_wmma_f32_16x16x32_f16(false, a, false, b, (short)0, c,
                                                false, false);
}

// A-matrix 16x32 f16 fragment: lane L (g = L>>4) holds row M = L&15,
// halves 0..7 -> K = k0 + 8g + i ; halves 8..15 -> K = k0 + 16 + 8g + (i-8)
__device__ __forceinline__ v16h load_a32(const half_t* __restrict__ p, int g) {
  union { v16h v; v8h h[2]; } u;
  u.h[0] = *(const v8h*)(p + 8 * g);
  u.h[1] = *(const v8h*)(p + 16 + 8 * g);
  return u.v;
}

// B-matrix 32x16 f16 fragment: lane L holds col N = L&15,
// halves 0..7 -> K = k0 + 8g + i ; halves 8..15 -> K = k0 + 16 + 8g + (i-8)
__device__ __forceinline__ v16h load_b32(const half_t* __restrict__ p, int g) {
  union { v16h v; v8h h[2]; } u;
  u.h[0] = *(const v8h*)(p + 8 * g);
  u.h[1] = *(const v8h*)(p + 16 + 8 * g);
  return u.v;
}

__global__ __launch_bounds__(256) void cvt_f32_f16(const float* __restrict__ s,
                                                   half_t* __restrict__ d, int n4) {
  int i = blockIdx.x * blockDim.x + threadIdx.x;
  if (i < n4) {
    const float4 f = ((const float4*)s)[i];
    v4h h;
    h[0] = (half_t)f.x; h[1] = (half_t)f.y; h[2] = (half_t)f.z; h[3] = (half_t)f.w;
    ((v4h*)d)[i] = h;
  }
}

// qkv = x @ W_qkv^T ; each wave: 64 token-rows x 64 out-cols (16 WMMA / k-step).
// Split into Q (pre-scaled 1/8), K ([B,H,S,HD]) and V^T ([B,H,HD,S]).
__global__ __launch_bounds__(128) void qkv_gemm_kernel(
    const half_t* __restrict__ xh, const half_t* __restrict__ wh,
    const float* __restrict__ bias,
    half_t* __restrict__ qb, half_t* __restrict__ kbuf, half_t* __restrict__ vtb) {
  const int lane = threadIdx.x & 31;
  const int wave = threadIdx.x >> 5;
  const int g = lane >> 4, ln = lane & 15;
  const int mt64 = blockIdx.y;                // 0..127 : 64 token rows
  const int nt64 = blockIdx.x * 4 + wave;     // 0..47  : 64 output cols
  const half_t* arow = xh + (size_t)(mt64 * 64 + ln) * DIN;        // + i*16*DIN
  const half_t* brow = wh + (size_t)(nt64 * 64 + ln) * DIN;        // + j*16*DIN
  v8f acc[4][4];
#pragma unroll
  for (int i = 0; i < 4; ++i)
#pragma unroll
    for (int j = 0; j < 4; ++j) acc[i][j] = (v8f){};
  for (int k0 = 0; k0 < DIN; k0 += 32) {
    __builtin_prefetch(arow + k0 + 512, 0, 0);   // stream-ahead on activations
    v16h af[4], bf[4];
#pragma unroll
    for (int i = 0; i < 4; ++i) af[i] = load_a32(arow + i * 16 * DIN + k0, g);
#pragma unroll
    for (int j = 0; j < 4; ++j) bf[j] = load_b32(brow + j * 16 * DIN + k0, g);
#pragma unroll
    for (int i = 0; i < 4; ++i)
#pragma unroll
      for (int j = 0; j < 4; ++j) acc[i][j] = wmma_f16(af[i], bf[j], acc[i][j]);
  }
  // Epilogue: lane holds col = nt64*64 + j*16 + ln, rows m = i*16 + r + 8g
#pragma unroll
  for (int j = 0; j < 4; ++j) {
    const int col = nt64 * 64 + j * 16 + ln;
    const float bv = bias[col];
    const int h = col / 192;
    const int which = (col >> 6) % 3;   // constant per 16-wide subtile
    const int d = col & 63;
#pragma unroll
    for (int i = 0; i < 4; ++i) {
#pragma unroll
      for (int r = 0; r < 8; ++r) {
        const int t = mt64 * 64 + i * 16 + r + 8 * g;
        const int bi = t >> 11, s = t & (Sq - 1);
        const float v = acc[i][j][r] + bv;
        const int bh = bi * Hh + h;
        if (which == 0)
          qb[(size_t)(bh * Sq + s) * HD + d] = (half_t)(v * 0.125f);
        else if (which == 1)
          kbuf[(size_t)(bh * Sq + s) * HD + d] = (half_t)v;
        else
          vtb[((size_t)bh * HD + d) * Sq + s] = (half_t)v;
      }
    }
  }
}

// Flash attention: one wave per (bh, pair of 16-row q tiles). K/V fragments are
// loaded once per 32-key chunk and reused for both q tiles (16 WMMA / chunk).
__global__ __launch_bounds__(128) void flash_attn_kernel(
    const half_t* __restrict__ qb, const half_t* __restrict__ kbuf,
    const half_t* __restrict__ vtb, half_t* __restrict__ ao) {
  const int lane = threadIdx.x & 31;
  const int wave = threadIdx.x >> 5;
  const int g = lane >> 4, ln = lane & 15;
  const int qp = blockIdx.x * 4 + wave;  // 0..63 : q-tile pair
  const int bh = blockIdx.y;             // 0..63

  const half_t* qbase = qb + (size_t)bh * Sq * HD;
  v16h qf0[2], qf1[2];
#pragma unroll
  for (int t = 0; t < 2; ++t) {
    const half_t* qrow = qbase + (size_t)((qp * 2 + t) * 16 + ln) * HD;
    qf0[t] = load_b32(qrow + 0, g);
    qf1[t] = load_b32(qrow + 32, g);
  }

  v8f o[2][4];
#pragma unroll
  for (int t = 0; t < 2; ++t)
#pragma unroll
    for (int dt = 0; dt < 4; ++dt) o[t][dt] = (v8f){};
  float mrun[2] = {-3.0e38f, -3.0e38f}, lrun[2] = {0.f, 0.f};
  const half_t* kbase = kbuf + (size_t)bh * Sq * HD;
  const half_t* vbase = vtb + (size_t)bh * HD * Sq;

  for (int key0 = 0; key0 < Sq; key0 += 32) {
    const half_t* krowA = kbase + (size_t)(key0 + ln) * HD;   // keys key0..+15
    const half_t* krowB = krowA + 16 * HD;                    // keys key0+16..+31
    const v16h ka0 = load_a32(krowA + 0, g);
    const v16h ka1 = load_a32(krowA + 32, g);
    const v16h kb0 = load_a32(krowB + 0, g);
    const v16h kb1 = load_a32(krowB + 32, g);
    v16h pv[2];
#pragma unroll
    for (int t = 0; t < 2; ++t) {
      v8f c1 = {}, c2 = {};
      c1 = wmma_f16(ka0, qf0[t], c1);
      c1 = wmma_f16(ka1, qf1[t], c1);
      c2 = wmma_f16(kb0, qf0[t], c2);
      c2 = wmma_f16(kb1, qf1[t], c2);
      // Lane holds, for q = ln: keys r+8g (c1) and 16+r+8g (c2).
      float mloc = c1[0];
#pragma unroll
      for (int r = 0; r < 8; ++r) { mloc = fmaxf(mloc, c1[r]); mloc = fmaxf(mloc, c2[r]); }
      mloc = fmaxf(mloc, __shfl_xor(mloc, 16, 32));
      const float mnew = fmaxf(mrun[t], mloc);
      const float alpha = __expf(mrun[t] - mnew);
      float p1[8], p2[8], ssum = 0.f;
#pragma unroll
      for (int r = 0; r < 8; ++r) {
        p1[r] = __expf(c1[r] - mnew);
        p2[r] = __expf(c2[r] - mnew);
        ssum += p1[r] + p2[r];
      }
      ssum += __shfl_xor(ssum, 16, 32);
      lrun[t] = lrun[t] * alpha + ssum;
      mrun[t] = mnew;
#pragma unroll
      for (int r = 0; r < 8; ++r) {
        const float ar = __shfl(alpha, r + 8 * g, 32);   // alpha of q-row r+8g
        o[t][0][r] *= ar; o[t][1][r] *= ar; o[t][2][r] *= ar; o[t][3][r] *= ar;
      }
      union { v16h v; half_t e[16]; } pf;
#pragma unroll
      for (int r = 0; r < 8; ++r) { pf.e[r] = (half_t)p1[r]; pf.e[8 + r] = (half_t)p2[r]; }
      pv[t] = pf.v;
    }
#pragma unroll
    for (int dt = 0; dt < 4; ++dt) {
      const v16h vf = load_b32(vbase + (size_t)(dt * 16 + ln) * Sq + key0, g);
      o[0][dt] = wmma_f16(pv[0], vf, o[0][dt]);
      o[1][dt] = wmma_f16(pv[1], vf, o[1][dt]);
    }
  }
  const int b = bh >> 4, h = bh & 15;
#pragma unroll
  for (int t = 0; t < 2; ++t) {
    const float linv = 1.0f / lrun[t];
#pragma unroll
    for (int r = 0; r < 8; ++r) {
      const float lr = __shfl(linv, r + 8 * g, 32);
      const size_t rowoff =
          ((size_t)b * Sq + (qp * 2 + t) * 16 + r + 8 * g) * EMB + h * HD;
      ao[rowoff + 0 * 16 + ln] = (half_t)(o[t][0][r] * lr);
      ao[rowoff + 1 * 16 + ln] = (half_t)(o[t][1][r] * lr);
      ao[rowoff + 2 * 16 + ln] = (half_t)(o[t][2][r] * lr);
      ao[rowoff + 3 * 16 + ln] = (half_t)(o[t][3][r] * lr);
    }
  }
}

// out = attn_out @ W_out^T + b_out ; each wave: 64 rows x 64 cols, f32 result.
__global__ __launch_bounds__(128) void out_gemm_kernel(
    const half_t* __restrict__ ah, const half_t* __restrict__ wh,
    const float* __restrict__ bias, float* __restrict__ out) {
  const int lane = threadIdx.x & 31, wave = threadIdx.x >> 5;
  const int g = lane >> 4, ln = lane & 15;
  const int mt64 = blockIdx.y;               // 0..127
  const int nt64 = blockIdx.x * 4 + wave;    // 0..15
  const half_t* arow = ah + (size_t)(mt64 * 64 + ln) * EMB;
  const half_t* brow = wh + (size_t)(nt64 * 64 + ln) * EMB;
  v8f acc[4][4];
#pragma unroll
  for (int i = 0; i < 4; ++i)
#pragma unroll
    for (int j = 0; j < 4; ++j) acc[i][j] = (v8f){};
  for (int k0 = 0; k0 < EMB; k0 += 32) {
    __builtin_prefetch(arow + k0 + 512, 0, 0);
    v16h af[4], bf[4];
#pragma unroll
    for (int i = 0; i < 4; ++i) af[i] = load_a32(arow + i * 16 * EMB + k0, g);
#pragma unroll
    for (int j = 0; j < 4; ++j) bf[j] = load_b32(brow + j * 16 * EMB + k0, g);
#pragma unroll
    for (int i = 0; i < 4; ++i)
#pragma unroll
      for (int j = 0; j < 4; ++j) acc[i][j] = wmma_f16(af[i], bf[j], acc[i][j]);
  }
#pragma unroll
  for (int j = 0; j < 4; ++j) {
    const int col = nt64 * 64 + j * 16 + ln;
    const float bv = bias[col];
#pragma unroll
    for (int i = 0; i < 4; ++i)
#pragma unroll
      for (int r = 0; r < 8; ++r)
        out[(size_t)(mt64 * 64 + i * 16 + r + 8 * g) * EMB + col] = acc[i][j][r] + bv;
  }
}

extern "C" void kernel_launch(void* const* d_in, const int* in_sizes, int n_in,
                              void* d_out, int out_size, void* d_ws, size_t ws_size,
                              hipStream_t stream) {
  const float* x    = (const float*)d_in[0];
  const float* Wqkv = (const float*)d_in[1];
  const float* bqkv = (const float*)d_in[2];
  const float* Wout = (const float*)d_in[3];
  const float* bout = (const float*)d_in[4];
  float* out = (float*)d_out;

  // Workspace layout (halves): needs ~88 MB.
  half_t* ws = (half_t*)d_ws;
  const size_t N_X  = (size_t)Bq * Sq * DIN;   // 8388608
  const size_t N_WQ = (size_t)3 * EMB * DIN;   // 3145728
  const size_t N_WO = (size_t)EMB * EMB;       // 1048576
  half_t* xh  = ws;
  half_t* wqh = xh + N_X;
  half_t* woh = wqh + N_WQ;
  half_t* qbf = woh + N_WO;
  half_t* kbf = qbf + N_X;
  half_t* vtb = kbf + N_X;
  half_t* aob = vtb + N_X;

  cvt_f32_f16<<<dim3((unsigned)((N_X / 4 + 255) / 256)), dim3(256), 0, stream>>>(x, xh, (int)(N_X / 4));
  cvt_f32_f16<<<dim3((unsigned)((N_WQ / 4 + 255) / 256)), dim3(256), 0, stream>>>(Wqkv, wqh, (int)(N_WQ / 4));
  cvt_f32_f16<<<dim3((unsigned)((N_WO / 4 + 255) / 256)), dim3(256), 0, stream>>>(Wout, woh, (int)(N_WO / 4));

  // QKV: 48 N64-tiles / 4 waves, 128 M64-tiles
  qkv_gemm_kernel<<<dim3(12, 128), dim3(128), 0, stream>>>(xh, wqh, bqkv, qbf, kbf, vtb);
  // Attention: 64 q-pairs / 4 waves, 64 (b,h) pairs
  flash_attn_kernel<<<dim3(16, 64), dim3(128), 0, stream>>>(qbf, kbf, vtb, aob);
  // Out-proj: 16 N64-tiles / 4 waves, 128 M64-tiles
  out_gemm_kernel<<<dim3(4, 128), dim3(128), 0, stream>>>(aob, woh, bout, out);
}
